// ExpertParallelMoE_59622736003407
// MI455X (gfx1250) — compile-verified, matrix operand order checked by reading, not run on hardware
//
#include <hip/hip_runtime.h>
#include <hip/hip_bf16.h>

// Problem constants (match reference)
#define B_  4
#define S_  2048
#define T_  (B_ * S_)      // 8192 tokens
#define H_  512
#define E_  16
#define F_  1024
#define TM_ 32                  // tokens per expert tile (M)
#define MAX_TILES_ (T_ / TM_)   // 256 tiles per expert worst case

// padded LDS row stride (bf16 elements) -> conflict-free b128 lane access
// buffer is [32][520] bf16 = 33,280 B; serves as x tile (32 x 512 cols of H)
// and as hid tile (32 rows x 512-col halves of F)
#define SXP_ 520

typedef __attribute__((ext_vector_type(16))) __bf16        v16bf;
typedef __attribute__((ext_vector_type(8)))  float         v8f;
typedef __attribute__((ext_vector_type(8)))  unsigned int  v8u;

__device__ __forceinline__ unsigned int f2bf(float f) {
    unsigned int u = __float_as_uint(f);
    unsigned int r = ((u >> 16) & 1u) + 0x7fffu;   // round-to-nearest-even
    return (u + r) >> 16;
}

__device__ __forceinline__ unsigned int pack_bf2(float lo, float hi) {
    return (f2bf(lo) & 0xffffu) | (f2bf(hi) << 16);
}

__device__ __forceinline__ int frag_k(int koff, int j) {
    return koff + ((j >> 2) << 4) + ((j & 3) << 1);
}

__device__ __forceinline__ v16bf ldsA(const unsigned short* p) {
    uint4 a0 = *(const uint4*)p;
    uint4 a1 = *(const uint4*)(p + 16);
    v8u ad;
    ad[0] = a0.x; ad[1] = a0.y; ad[2] = a0.z; ad[3] = a0.w;
    ad[4] = a1.x; ad[5] = a1.y; ad[6] = a1.z; ad[7] = a1.w;
    return __builtin_bit_cast(v16bf, ad);
}

// ---------------------------------------------------------------------------
// Kernel 0: zero output accumulator + expert counters
// ---------------------------------------------------------------------------
__global__ void zero_kernel(float* __restrict__ out, int n, int* __restrict__ cnt) {
    int i = blockIdx.x * blockDim.x + threadIdx.x;
    if (i < n) out[i] = 0.0f;
    if (blockIdx.x == 0 && threadIdx.x < E_) cnt[threadIdx.x] = 0;
}

// ---------------------------------------------------------------------------
// Kernel W: pre-pack weight tensor [E, K, N] fp32 -> bf16 WMMA B-fragments.
// frag_id = (e*(K/32)+kc)*(N/16)+ntile ; per lane 8 contiguous dwords (1KB/frag)
// grid.x = E * (K/32) * (N/128), block = 256
// ---------------------------------------------------------------------------
__global__ __launch_bounds__(256)
void prepack_kernel(const float* __restrict__ W, unsigned int* __restrict__ P,
                    int K, int N) {
    const int nb = N >> 7;
    const int kb = K >> 5;
    int bid = blockIdx.x;
    int e   = bid / (kb * nb);
    int r0  = bid % (kb * nb);
    int kc  = r0 / nb;
    int cb  = r0 % nb;

    __shared__ float s_w[32 * 128];        // 16 KB fp32 tile
    const float* We = W + (size_t)e * K * N + (size_t)(kc * 32) * N + cb * 128;
    {
        int r = threadIdx.x >> 3;
        int q = threadIdx.x & 7;
        const float4* src = (const float4*)(We + (size_t)r * N + q * 16);
        float4* dst = (float4*)&s_w[r * 128 + q * 16];
#pragma unroll
        for (int i = 0; i < 4; ++i) dst[i] = src[i];
    }
    __syncthreads();

    int ntl  = threadIdx.x >> 5;
    int lane = threadIdx.x & 31;
    int n    = lane & 15;
    int koff = (lane >> 4) * 8;

    v8u d;
#pragma unroll
    for (int j = 0; j < 8; ++j) {
        int k = frag_k(koff, j);
        d[j] = pack_bf2(s_w[k * 128 + ntl * 16 + n],
                        s_w[(k + 1) * 128 + ntl * 16 + n]);
    }
    size_t fragid = (size_t)(e * kb + kc) * (N >> 4) + (size_t)(cb * 8 + ntl);
    *(v8u*)(P + fragid * 256 + lane * 8) = d;
}

// ---------------------------------------------------------------------------
// Kernel 1: router logits, top-2, softmax, bucket tokens per expert
// ---------------------------------------------------------------------------
__global__ __launch_bounds__(256)
void router_topk_kernel(const float* __restrict__ x,
                        const float* __restrict__ router_w,
                        int*   __restrict__ cnt,
                        int*   __restrict__ tok,
                        float* __restrict__ tw) {
    __shared__ float s_rw[E_ * H_];                          // 32 KB
    for (int i = threadIdx.x; i < E_ * H_; i += 256) s_rw[i] = router_w[i];
    __syncthreads();

    int t = blockIdx.x * 256 + threadIdx.x;
    if (t >= T_) return;

    float acc[E_];
#pragma unroll
    for (int e = 0; e < E_; ++e) acc[e] = 0.0f;

    const float4* xr = (const float4*)(x + (size_t)t * H_);
    for (int h4 = 0; h4 < H_ / 4; ++h4) {
        float4 xv = xr[h4];
        int h = h4 * 4;
#pragma unroll
        for (int e = 0; e < E_; ++e) {
            acc[e] += xv.x * s_rw[e * H_ + h + 0]
                    + xv.y * s_rw[e * H_ + h + 1]
                    + xv.z * s_rw[e * H_ + h + 2]
                    + xv.w * s_rw[e * H_ + h + 3];
        }
    }

    int i0 = 0; float v0 = acc[0];
#pragma unroll
    for (int e = 1; e < E_; ++e) { if (acc[e] > v0) { v0 = acc[e]; i0 = e; } }
    int i1 = -1; float v1 = -3.4e38f;
#pragma unroll
    for (int e = 0; e < E_; ++e) { if (e != i0 && acc[e] > v1) { v1 = acc[e]; i1 = e; } }

    float w0 = 1.0f / (1.0f + __expf(v1 - v0));
    float w1 = 1.0f - w0;

    int p0 = atomicAdd(&cnt[i0], 1);
    tok[i0 * T_ + p0] = t;  tw[i0 * T_ + p0] = w0;
    int p1 = atomicAdd(&cnt[i1], 1);
    tok[i1 * T_ + p1] = t;  tw[i1 * T_ + p1] = w1;
}

// ---------------------------------------------------------------------------
// Kernel 2: per-expert SwiGLU MLP on gathered 32-token tiles, WMMA bf16.
// Gate/up: B fragments reused across 2 M-subtiles.  hid kept as packed-bf16
// registers; down-GEMM runs in two K-phases (F columns 0..511 then 512..1023)
// through the same 33 KB LDS buffer, so every down B fragment is read once
// and also reused across 2 M-subtiles.
// grid.x = E_ * MAX_TILES_, block = 256 (8 waves)
// ---------------------------------------------------------------------------
__global__ __launch_bounds__(256)
void expert_mlp_kernel(const float* __restrict__ x,
                       const unsigned int* __restrict__ Gp,
                       const unsigned int* __restrict__ Up,
                       const unsigned int* __restrict__ Dp,
                       const int*   __restrict__ cnt,
                       const int*   __restrict__ tok,
                       const float* __restrict__ tw,
                       float*       __restrict__ out) {
    const int e    = blockIdx.x / MAX_TILES_;
    const int tile = blockIdx.x % MAX_TILES_;
    const int ne   = cnt[e];
    if (tile * TM_ >= ne) return;   // uniform: whole block exits

    __shared__ __align__(16) unsigned short s_xh[TM_ * SXP_];   // 33,280 B
    __shared__ int   s_tok[TM_];
    __shared__ float s_wt[TM_];

    if (threadIdx.x < TM_) {
        int idx = tile * TM_ + threadIdx.x;
        if (idx < ne) {
            s_tok[threadIdx.x] = tok[e * T_ + idx];
            s_wt[threadIdx.x]  = tw[e * T_ + idx];
        } else {
            s_tok[threadIdx.x] = tok[e * T_];
            s_wt[threadIdx.x]  = 0.0f;
        }
    }
    __syncthreads();

    // ---- gather x rows -> bf16 LDS tile [32][H_] -----------------------
    {
        int r  = threadIdx.x >> 3;          // 0..31 row
        int c0 = (threadIdx.x & 7) * 64;    // 64 cols per thread
        const float4* xr4 = (const float4*)(x + (size_t)s_tok[r] * H_ + c0);
#pragma unroll
        for (int i = 0; i < 16; ++i) {
            float4 v = xr4[i];
            uint2 pk;
            pk.x = pack_bf2(v.x, v.y);
            pk.y = pack_bf2(v.z, v.w);
            *(uint2*)&s_xh[r * SXP_ + c0 + i * 4] = pk;
        }
    }
    __syncthreads();

    const int lane = threadIdx.x & 31;
    const int wv   = threadIdx.x >> 5;            // wave 0..7
    const int m    = lane & 15;
    const int koff = (lane >> 4) * 8;
    const int mrow = (lane >> 4) ? 8 : 0;

    const unsigned int* Ge = Gp + (size_t)e * ((size_t)H_ * F_ / 2);
    const unsigned int* Ue = Up + (size_t)e * ((size_t)H_ * F_ / 2);
    const unsigned int* De = Dp + (size_t)e * ((size_t)F_ * H_ / 2);

    __builtin_prefetch(Ge + (size_t)(wv * 8) * 256 + lane * 8, 0, 1);
    __builtin_prefetch(Ue + (size_t)(wv * 8) * 256 + lane * 8, 0, 1);

    // ---- GEMM 1+2: gate & up, [32,H] x [H,128-per-wave] ----------------
    v8f accg[2][8], accu[2][8];
#pragma unroll
    for (int mt = 0; mt < 2; ++mt)
#pragma unroll
        for (int nt = 0; nt < 8; ++nt)
#pragma unroll
            for (int r = 0; r < 8; ++r) { accg[mt][nt][r] = 0.0f; accu[mt][nt][r] = 0.0f; }

    for (int kc = 0; kc < H_ / 32; ++kc) {
        const int k0 = kc * 32;
        v16bf av0 = ldsA(&s_xh[(m)      * SXP_ + k0 + koff]);
        v16bf av1 = ldsA(&s_xh[(m + 16) * SXP_ + k0 + koff]);

        const size_t fbase = ((size_t)kc * (F_ / 16) + wv * 8) * 256 + lane * 8;
#pragma unroll
        for (int nt = 0; nt < 8; ++nt) {
            v16bf bg = __builtin_bit_cast(v16bf, *(const v8u*)(Ge + fbase + (size_t)nt * 256));
            v16bf bu = __builtin_bit_cast(v16bf, *(const v8u*)(Ue + fbase + (size_t)nt * 256));
            accg[0][nt] = __builtin_amdgcn_wmma_f32_16x16x32_bf16(
                false, av0, false, bg, (short)0, accg[0][nt], false, false);
            accg[1][nt] = __builtin_amdgcn_wmma_f32_16x16x32_bf16(
                false, av1, false, bg, (short)0, accg[1][nt], false, false);
            accu[0][nt] = __builtin_amdgcn_wmma_f32_16x16x32_bf16(
                false, av0, false, bu, (short)0, accu[0][nt], false, false);
            accu[1][nt] = __builtin_amdgcn_wmma_f32_16x16x32_bf16(
                false, av1, false, bu, (short)0, accu[1][nt], false, false);
        }
    }

    // ---- hid = silu(gate)*up, packed bf16 in registers -----------------
    // (frees the 256 fp32 accumulators before the down-GEMM)
    unsigned int hp[2][8][4];
#pragma unroll
    for (int mt = 0; mt < 2; ++mt)
#pragma unroll
        for (int nt = 0; nt < 8; ++nt)
#pragma unroll
            for (int j = 0; j < 4; ++j) {
                float g0 = accg[mt][nt][2 * j],     u0 = accu[mt][nt][2 * j];
                float g1 = accg[mt][nt][2 * j + 1], u1 = accu[mt][nt][2 * j + 1];
                float h0 = (g0 / (1.0f + __expf(-g0))) * u0;
                float h1 = (g1 / (1.0f + __expf(-g1))) * u1;
                hp[mt][nt][j] = pack_bf2(h0, h1);
            }

    // ---- down-GEMM in two K-phases through the shared LDS buffer -------
    const int colb2 = wv * 64;
    v8f accd[2][4];
#pragma unroll
    for (int mt = 0; mt < 2; ++mt)
#pragma unroll
        for (int nt = 0; nt < 4; ++nt)
#pragma unroll
            for (int r = 0; r < 8; ++r) accd[mt][nt][r] = 0.0f;

#pragma unroll
    for (int p = 0; p < 2; ++p) {
        __syncthreads();   // previous readers of s_xh are done

        // waves owning F-columns [p*512, p*512+512) publish their hid slice
        if ((wv >> 2) == p) {
            int lc0 = (wv & 3) * 128;
#pragma unroll
            for (int mt = 0; mt < 2; ++mt)
#pragma unroll
                for (int nt = 0; nt < 8; ++nt) {
                    int col = lc0 + nt * 16 + m;
#pragma unroll
                    for (int j = 0; j < 4; ++j) {
                        unsigned int v = hp[mt][nt][j];
                        int row = mt * 16 + 2 * j + mrow;
                        s_xh[row * SXP_ + col]       = (unsigned short)(v & 0xffffu);
                        s_xh[(row + 1) * SXP_ + col] = (unsigned short)(v >> 16);
                    }
                }
        }
        __syncthreads();

        // accumulate down-GEMM over this K-phase: kc = p*16 .. p*16+15
        for (int kcl = 0; kcl < 16; ++kcl) {
            const int kc = p * 16 + kcl;
            const int k0 = kcl * 32;
            v16bf av0 = ldsA(&s_xh[(m)      * SXP_ + k0 + koff]);
            v16bf av1 = ldsA(&s_xh[(m + 16) * SXP_ + k0 + koff]);
            const size_t fbase = ((size_t)kc * (H_ / 16) + wv * 4) * 256 + lane * 8;
#pragma unroll
            for (int nt = 0; nt < 4; ++nt) {
                v16bf bd = __builtin_bit_cast(v16bf, *(const v8u*)(De + fbase + (size_t)nt * 256));
                accd[0][nt] = __builtin_amdgcn_wmma_f32_16x16x32_bf16(
                    false, av0, false, bd, (short)0, accd[0][nt], false, false);
                accd[1][nt] = __builtin_amdgcn_wmma_f32_16x16x32_bf16(
                    false, av1, false, bd, (short)0, accd[1][nt], false, false);
            }
        }
    }

    // ---- weighted scatter-add into output ------------------------------
#pragma unroll
    for (int mt = 0; mt < 2; ++mt)
#pragma unroll
        for (int nt = 0; nt < 4; ++nt) {
            int col = colb2 + nt * 16 + m;
#pragma unroll
            for (int r = 0; r < 8; ++r) {
                int mm = mt * 16 + r + mrow;
                float val = s_wt[mm] * accd[mt][nt][r];
                atomicAdd(&out[(size_t)s_tok[mm] * H_ + col], val);
            }
        }
}

// ---------------------------------------------------------------------------
// Host launcher
// ---------------------------------------------------------------------------
extern "C" void kernel_launch(void* const* d_in, const int* in_sizes, int n_in,
                              void* d_out, int out_size, void* d_ws, size_t ws_size,
                              hipStream_t stream) {
    const float* x        = (const float*)d_in[0];   // [T,H]
    const float* router_w = (const float*)d_in[1];   // [E,H]
    const float* gate_w   = (const float*)d_in[2];   // [E,H,F]
    const float* up_w     = (const float*)d_in[3];   // [E,H,F]
    const float* down_w   = (const float*)d_in[4];   // [E,F,H]
    float*       out      = (float*)d_out;           // [T,H]

    char* ws = (char*)d_ws;
    int*   cnt = (int*)ws;                                        // E ints
    int*   tok = (int*)(ws + 256);                                // E*T ints
    float* twp = (float*)(ws + 256 + (size_t)E_ * T_ * 4);        // E*T floats
    size_t packed_off = 256 + (size_t)E_ * T_ * 8;                // ~1 MB
    unsigned int* Gp = (unsigned int*)(ws + packed_off);          // 16 MB
    unsigned int* Up = Gp + (size_t)E_ * H_ * F_ / 2;             // 16 MB
    unsigned int* Dp = Up + (size_t)E_ * H_ * F_ / 2;             // 16 MB

    {
        int n = T_ * H_;
        zero_kernel<<<(n + 255) / 256, 256, 0, stream>>>(out, n, cnt);
    }
    prepack_kernel<<<E_ * (H_ / 32) * (F_ / 128), 256, 0, stream>>>(gate_w, Gp, H_, F_);
    prepack_kernel<<<E_ * (H_ / 32) * (F_ / 128), 256, 0, stream>>>(up_w,   Up, H_, F_);
    prepack_kernel<<<E_ * (F_ / 32) * (H_ / 128), 256, 0, stream>>>(down_w, Dp, F_, H_);
    router_topk_kernel<<<T_ / 256, 256, 0, stream>>>(x, router_w, cnt, tok, twp);
    expert_mlp_kernel<<<E_ * MAX_TILES_, 256, 0, stream>>>(
        x, Gp, Up, Dp, cnt, tok, twp, out);
}
